// TransformerDecoderLayer_70574902608254
// MI455X (gfx1250) — matrix operand order repaired and checked by reading, not compiled, not run on hardware
//
#include <hip/hip_runtime.h>

namespace {

constexpr int   kC    = 256;
constexpr int   kB    = 8;
constexpr int   kNQ   = 900;
constexpr int   kRows = kB * kNQ;   // 7200
constexpr int   kH    = 8;
constexpr int   kDH   = 32;
constexpr int   kLV   = 21760;
constexpr int   kVTP  = 928;        // V^T padded K extent (29*32)
constexpr float kEps  = 1e-5f;
constexpr float kScale = 0.1767766953f;   // 1/sqrt(32)

typedef unsigned short u16;
typedef __attribute__((ext_vector_type(16))) __bf16 v16bf;
typedef __attribute__((ext_vector_type(8)))  float  v8f;
typedef __attribute__((ext_vector_type(4)))  float  f32x4;
typedef __attribute__((ext_vector_type(4)))  unsigned int u32x4;
typedef __attribute__((ext_vector_type(2)))  unsigned int u32x2;
typedef __attribute__((ext_vector_type(4)))  int i32x4;
typedef __attribute__((ext_vector_type(8)))  int i32x8;

union BF16Frag { v16bf v; u16 u[16]; u32x4 q[2]; };

__device__ __forceinline__ u16 f2bf(float f) {
  unsigned int u = __float_as_uint(f);
  u += 0x7FFFu + ((u >> 16) & 1u);            // round-to-nearest-even
  return (u16)(u >> 16);
}
__device__ __forceinline__ float bf2f(u16 h) {
  return __uint_as_float(((unsigned int)h) << 16);
}

// ---------------- packed f32 -> bf16 cast (4 elements / thread, optional scale) ------------
__global__ __launch_bounds__(256) void cast_bf16(const float* __restrict__ in,
                                                 u16* __restrict__ outp,
                                                 int n4, float scale) {
  const int i = blockIdx.x * 256 + threadIdx.x;
  if (i >= n4) return;
  const f32x4 f = ((const f32x4*)in)[i];
  u32x2 r;
  r.x = (unsigned)f2bf(f.x * scale) | ((unsigned)f2bf(f.y * scale) << 16);
  r.y = (unsigned)f2bf(f.z * scale) | ((unsigned)f2bf(f.w * scale) << 16);
  ((u32x2*)outp)[i] = r;
}

// ---------------- bias scale helper (fold 1/sqrt(d) into wq bias) ----------------
__global__ __launch_bounds__(256) void scale_bias(const float* __restrict__ in,
                                                  float* __restrict__ outp,
                                                  int n, int nscale, float s) {
  const int i = blockIdx.x * 256 + threadIdx.x;
  if (i < n) outp[i] = in[i] * (i < nscale ? s : 1.f);
}

// ---------------- elementwise add ----------------
__global__ __launch_bounds__(256) void ew_add(const float* __restrict__ a,
                                              const float* __restrict__ b,
                                              float* __restrict__ c, int n) {
  int i = blockIdx.x * 256 + threadIdx.x;
  if (i < n) c[i] = a[i] + b[i];
}

// ---------------- TDM: 2D tile load global(bf16) -> LDS ----------------
// D# group0: [1:0]=count=1, [63:32]=lds_addr, [120:64]=global_addr, [127:126]=type=2
// D# group1: [17:16]=data_size(1=2B), [79:48]=tensor_dim0, [111:80]=tensor_dim1,
//            [127:112]=tile_dim0, [143:128]=tile_dim1, [207:160]=tensor_dim0_stride
__device__ __forceinline__ void tdm_load_2d(unsigned lds_addr, const void* gptr,
                                            unsigned d0, unsigned d1, unsigned strideElems) {
  const unsigned long long ga = (unsigned long long)(uintptr_t)gptr;
  u32x4 g0;
  g0.x = 1u;                                        // count=1 (user descriptor)
  g0.y = lds_addr;
  g0.z = (unsigned)(ga & 0xFFFFFFFFu);
  g0.w = (unsigned)((ga >> 32) & 0x01FFFFFFu) | (2u << 30);   // type=2 ("image")
  i32x8 g1;
  g1[0] = (int)(1u << 16);                          // data_size = 2 bytes
  g1[1] = (int)((d0 & 0xFFFFu) << 16);              // tensor_dim0[15:0]
  g1[2] = (int)(((d0 >> 16) & 0xFFFFu) | ((d1 & 0xFFFFu) << 16));
  g1[3] = (int)(((d1 >> 16) & 0xFFFFu) | ((d0 & 0xFFFFu) << 16));  // tile_dim0 = d0
  g1[4] = (int)(d1 & 0xFFFFu);                      // tile_dim1 = d1
  g1[5] = (int)strideElems;                         // tensor_dim0_stride[31:0]
  g1[6] = 0;
  g1[7] = 0;
  i32x4 gz4 = {0, 0, 0, 0};
  i32x8 gz8 = {0, 0, 0, 0, 0, 0, 0, 0};
  __builtin_amdgcn_tensor_load_to_lds(g0, g1, gz4, gz4, gz8, 0);
}

// ---------------- TDM double-buffered WMMA GEMM ----------------
// C[M,N] = A[M,K](bf16) @ W[N,K](bf16)^T + bias ; OBF: bf16 store ; RELU fused.
// Block: 8 waves = 128 rows x 16 cols. Each wave DMA's its own 16x32 A slice and
// a private copy of the 16x32 W tile into LDS, double buffered on TENSORcnt.
template <bool OBF, bool RELU>
__global__ __launch_bounds__(256) void gemm_tdm(
    const u16* __restrict__ A, const u16* __restrict__ W,
    const float* __restrict__ bias, void* __restrict__ Cp,
    int M, int N, int K) {
  __shared__ u16 Abuf[2][128 * 32];      // 16 KB
  __shared__ u16 Wbuf[2][8][16 * 32];    // 16 KB
  float* Cf = (float*)Cp;
  u16*   Cb = (u16*)Cp;

  const int lane = threadIdx.x & 31;
  const int wave = threadIdx.x >> 5;
  const int row0 = blockIdx.x * 128 + wave * 16;
  if (row0 >= M) return;                            // wave-uniform exit
  const int col0 = blockIdx.y * 16;
  const int l15 = lane & 15, g = lane >> 4;
  const int kbA = g * 16, kbB = g * 16;

  const u16* Ag = A + (size_t)row0 * K;
  const u16* Wg = W + (size_t)col0 * K;
  const unsigned aLds0 = (unsigned)(uintptr_t)&Abuf[0][wave * 512];
  const unsigned aLds1 = (unsigned)(uintptr_t)&Abuf[1][wave * 512];
  const unsigned wLds0 = (unsigned)(uintptr_t)&Wbuf[0][wave][0];
  const unsigned wLds1 = (unsigned)(uintptr_t)&Wbuf[1][wave][0];

  const int nK = K >> 5;
  // prologue: chunk 0 -> buffer 0
  tdm_load_2d(aLds0, Ag, 32, 16, (unsigned)K);
  tdm_load_2d(wLds0, Wg, 32, 16, (unsigned)K);

  v8f acc = {};
  for (int kc = 0; kc < nK; ++kc) {
    const int buf = kc & 1;
    if (kc + 1 < nK) {                              // issue next chunk into other buffer
      const size_t ko = (size_t)(kc + 1) << 5;
      asm volatile("" ::: "memory");
      tdm_load_2d(buf ? aLds0 : aLds1, Ag + ko, 32, 16, (unsigned)K);
      tdm_load_2d(buf ? wLds0 : wLds1, Wg + ko, 32, 16, (unsigned)K);
      __builtin_amdgcn_s_wait_tensorcnt(2);         // current chunk's pair has landed
    } else {
      __builtin_amdgcn_s_wait_tensorcnt(0);
    }
    asm volatile("" ::: "memory");

    const u16* As = &Abuf[buf][wave * 512];
    const u16* Ws = &Wbuf[buf][wave][0];
    BF16Frag a, b;
    a.q[0] = *(const u32x4*)(As + l15 * 32 + kbA);        // K = kbA..kbA+7
    a.q[1] = *(const u32x4*)(As + l15 * 32 + kbA + 8);    // K = kbA+8..+15
    b.q[0] = *(const u32x4*)(Ws + l15 * 32 + kbB);        // K = kbB..kbB+7
    b.q[1] = *(const u32x4*)(Ws + l15 * 32 + kbB + 8);    // K = kbB+8..+15
    acc = __builtin_amdgcn_wmma_f32_16x16x32_bf16(false, a.v, false, b.v,
                                                  (short)0, acc, false, false);
  }

  const int nCol = col0 + l15;
  const float bn = bias ? bias[nCol] : 0.f;
#pragma unroll
  for (int i = 0; i < 8; ++i) {
    const int m = row0 + i + 8 * g;
    float v = acc[i] + bn;
    if (RELU) v = fmaxf(v, 0.f);
    if (OBF) Cb[(size_t)m * N + nCol] = f2bf(v);
    else     Cf[(size_t)m * N + nCol] = v;
  }
}

// ---------------- split QKV (all bf16; scale already folded into wq) ----------------
// Qh,Kh: [B,H,900,32] ; VT: [B,H,32,928] (K-padded, pad zeroed by vt_zero)
__global__ __launch_bounds__(256) void qkv_split_bf(
    const u16* __restrict__ qk,   // [7200,512] bf16 (wq out | wk out)
    const u16* __restrict__ vpr,  // [7200,256] bf16
    u16* __restrict__ Qh, u16* __restrict__ Kh, u16* __restrict__ VT) {
  const int idx = blockIdx.x * 256 + threadIdx.x;
  if (idx >= kRows * kC) return;
  const int d = idx & 31;
  const int h = (idx >> 5) & 7;
  const int row = idx >> 8;
  const int b = row / kNQ, q = row % kNQ;
  const size_t o = (((size_t)(b * kH + h)) * kNQ + q) * kDH + d;
  Qh[o] = qk[(size_t)row * 512 + h * kDH + d];
  Kh[o] = qk[(size_t)row * 512 + 256 + h * kDH + d];
  VT[(((size_t)(b * kH + h)) * kDH + d) * kVTP + q] = vpr[(size_t)row * kC + h * kDH + d];
}

__global__ __launch_bounds__(256) void vt_zero(u16* __restrict__ VT) {
  const int idx = blockIdx.x * 256 + threadIdx.x;       // B*H*32*(928-900)
  const int nPad = kVTP - kNQ;
  if (idx >= kB * kH * kDH * nPad) return;
  const int q = kNQ + idx % nPad;
  const int r = idx / nPad;
  VT[(size_t)r * kVTP + q] = 0;
}

// ---------------- flash-style attention (bf16 operands): one wave per 16-query tile --------
__global__ __launch_bounds__(32) void attn_flash(
    const u16* __restrict__ Qh, const u16* __restrict__ Kh,
    const u16* __restrict__ VT, u16* __restrict__ O /* [B,NQ,256] bf16 */) {
  __shared__ u16 pstage[16 * 32];
  const int mt = blockIdx.x, h = blockIdx.y, b = blockIdx.z;
  const u16* Q  = Qh + ((size_t)(b * kH + h)) * kNQ * kDH;
  const u16* Km = Kh + ((size_t)(b * kH + h)) * kNQ * kDH;
  const u16* Vt = VT + ((size_t)(b * kH + h)) * kDH * kVTP;
  const int lane = threadIdx.x;
  const int l15 = lane & 15, g = lane >> 4;
  const int kbA = g * 16, kbB = g * 16;

  BF16Frag aq;
#pragma unroll
  for (int j = 0; j < 16; ++j) aq.u[j] = 0;
  {
    const int m = mt * 16 + l15;
    if (m < kNQ) {
      const u16* Qrow = Q + m * kDH;
      aq.q[0] = *(const u32x4*)(Qrow + kbA);
      aq.q[1] = *(const u32x4*)(Qrow + kbA + 8);
    }
  }

  // ---- pass 1: row max ----
  float mx[8];
#pragma unroll
  for (int i = 0; i < 8; ++i) mx[i] = -1e30f;
  const int nCT = (kNQ + 15) / 16;  // 57
  for (int ct = 0; ct < nCT; ++ct) {
    BF16Frag bk;
#pragma unroll
    for (int j = 0; j < 16; ++j) bk.u[j] = 0;
    const int col = ct * 16 + l15;
    if (col < kNQ) {
      const u16* Krow = Km + col * kDH;
      bk.q[0] = *(const u32x4*)(Krow + kbB);
      bk.q[1] = *(const u32x4*)(Krow + kbB + 8);
    }
    v8f s = {};
    s = __builtin_amdgcn_wmma_f32_16x16x32_bf16(false, aq.v, false, bk.v,
                                                (short)0, s, false, false);
    if (col < kNQ) {
#pragma unroll
      for (int i = 0; i < 8; ++i) mx[i] = fmaxf(mx[i], s[i]);
    }
  }
#pragma unroll
  for (int i = 0; i < 8; ++i) {
    float mm = mx[i];
    for (int d = 1; d < 16; d <<= 1) mm = fmaxf(mm, __shfl_xor(mm, d, 32));
    mx[i] = mm;
  }

  // ---- pass 2: exp, sum and P @ V ----
  float sum[8];
#pragma unroll
  for (int i = 0; i < 8; ++i) sum[i] = 0.f;
  v8f o0 = {}, o1 = {};
  const int nC2 = (kNQ + 31) / 32;  // 29
  for (int c2 = 0; c2 < nC2; ++c2) {
#pragma unroll
    for (int sub = 0; sub < 2; ++sub) {
      const int ct = c2 * 2 + sub;
      float p[8];
      if (ct < nCT) {
        BF16Frag bk;
#pragma unroll
        for (int j = 0; j < 16; ++j) bk.u[j] = 0;
        const int col = ct * 16 + l15;
        if (col < kNQ) {
          const u16* Krow = Km + col * kDH;
          bk.q[0] = *(const u32x4*)(Krow + kbB);
          bk.q[1] = *(const u32x4*)(Krow + kbB + 8);
        }
        v8f s = {};
        s = __builtin_amdgcn_wmma_f32_16x16x32_bf16(false, aq.v, false, bk.v,
                                                    (short)0, s, false, false);
#pragma unroll
        for (int i = 0; i < 8; ++i)
          p[i] = (col < kNQ) ? __expf(s[i] - mx[i]) : 0.f;
      } else {
#pragma unroll
        for (int i = 0; i < 8; ++i) p[i] = 0.f;
      }
#pragma unroll
      for (int i = 0; i < 8; ++i) {
        sum[i] += p[i];
        pstage[(i + 8 * g) * 32 + sub * 16 + l15] = f2bf(p[i]);   // C-layout -> LDS (bf16)
      }
    }
    __syncthreads();
    BF16Frag ap;                                            // LDS -> A-layout (b128 reads)
    ap.q[0] = *(const u32x4*)(pstage + l15 * 32 + kbA);
    ap.q[1] = *(const u32x4*)(pstage + l15 * 32 + kbA + 8);
    __syncthreads();

    // V^T rows are zero-padded beyond K=900: no guards needed.
    BF16Frag bv0, bv1;
    const u16* vt0 = Vt + (size_t)l15 * kVTP + c2 * 32 + kbB;
    const u16* vt1 = Vt + (size_t)(16 + l15) * kVTP + c2 * 32 + kbB;
    bv0.q[0] = *(const u32x4*)(vt0);
    bv0.q[1] = *(const u32x4*)(vt0 + 8);
    bv1.q[0] = *(const u32x4*)(vt1);
    bv1.q[1] = *(const u32x4*)(vt1 + 8);
    o0 = __builtin_amdgcn_wmma_f32_16x16x32_bf16(false, ap.v, false, bv0.v,
                                                 (short)0, o0, false, false);
    o1 = __builtin_amdgcn_wmma_f32_16x16x32_bf16(false, ap.v, false, bv1.v,
                                                 (short)0, o1, false, false);
  }

#pragma unroll
  for (int i = 0; i < 8; ++i) {
    float s = sum[i];
    for (int d = 1; d < 16; d <<= 1) s += __shfl_xor(s, d, 32);
    sum[i] = s;
  }
#pragma unroll
  for (int i = 0; i < 8; ++i) {
    const int grow = mt * 16 + i + 8 * g;
    if (grow < kNQ) {
      const float inv = 1.f / sum[i];
      u16* orow = O + ((size_t)(b * kNQ + grow)) * kC + h * kDH;
      orow[l15]      = f2bf(o0[i] * inv);
      orow[16 + l15] = f2bf(o1[i] * inv);
    }
  }
}

// ---------------- fused residual-add + LayerNorm (one wave per 256-wide row) ----------------
__global__ __launch_bounds__(256) void add_ln(
    const float* __restrict__ x, const float* __restrict__ res,
    const float* __restrict__ gamma, const float* __restrict__ beta,
    float* __restrict__ out, const float* __restrict__ pos,
    float* __restrict__ out_pp, int rows) {
  const int row = blockIdx.x * 8 + (threadIdx.x >> 5);
  if (row >= rows) return;
  const int lane = threadIdx.x & 31;
  const float* xr = x + (size_t)row * kC;
  const float* rr = res + (size_t)row * kC;
  float v[8]; float s = 0.f;
#pragma unroll
  for (int i = 0; i < 8; ++i) { v[i] = xr[lane * 8 + i] + rr[lane * 8 + i]; s += v[i]; }
  for (int d = 1; d < 32; d <<= 1) s += __shfl_xor(s, d, 32);
  const float mean = s * (1.f / kC);
  float s2 = 0.f;
#pragma unroll
  for (int i = 0; i < 8; ++i) { const float dd = v[i] - mean; s2 += dd * dd; }
  for (int d = 1; d < 32; d <<= 1) s2 += __shfl_xor(s2, d, 32);
  const float rstd = rsqrtf(s2 * (1.f / kC) + kEps);
#pragma unroll
  for (int i = 0; i < 8; ++i) {
    const int c = lane * 8 + i;
    const float y = (v[i] - mean) * rstd * gamma[c] + beta[c];
    out[(size_t)row * kC + c] = y;
    if (out_pp) out_pp[(size_t)row * kC + c] = y + pos[(size_t)row * kC + c];
  }
}

// ---------------- softmax over the 16 (level,point) weights per head ----------------
__global__ __launch_bounds__(256) void aw_softmax(const float* __restrict__ in,
                                                  float* __restrict__ outp, int n) {
  const int t = blockIdx.x * 256 + threadIdx.x;
  if (t >= n) return;
  const int row = t >> 3, h = t & 7;
  const float* p = in + (size_t)row * 128 + h * 16;
  float* q = outp + (size_t)row * 128 + h * 16;
  float m = -1e30f; float v[16];
#pragma unroll
  for (int i = 0; i < 16; ++i) { v[i] = p[i]; m = fmaxf(m, v[i]); }
  float s = 0.f;
#pragma unroll
  for (int i = 0; i < 16; ++i) { v[i] = __expf(v[i] - m); s += v[i]; }
  const float inv = 1.f / s;
#pragma unroll
  for (int i = 0; i < 16; ++i) q[i] = v[i] * inv;
}

// ---------------- deformable bilinear sampling: one wave per (b,q,h), lane = channel --------
__global__ __launch_bounds__(256) void deform_sample(
    const u16* __restrict__ value,             // [B*LV, 256] bf16
    const float* __restrict__ offv,            // [7200,256]
    const float* __restrict__ aws,             // [7200,128] softmaxed
    const float* __restrict__ refp,            // [B,900,4,2]
    float* __restrict__ outp) {                // [7200,256]
  const int wid = blockIdx.x * 8 + (threadIdx.x >> 5);
  if (wid >= kRows * kH) return;
  const int h = wid & 7;
  const int row = wid >> 3;
  const int b = row / kNQ;
  const int lane = threadIdx.x & 31;
  const int Hs[4]     = {128, 64, 32, 16};
  const int starts[4] = {0, 16384, 20480, 21504};
  float acc = 0.f;
  for (int l = 0; l < 4; ++l) {
    const int Wl = Hs[l], Hl = Hs[l];
    const float rx = refp[((size_t)row * 4 + l) * 2 + 0];
    const float ry = refp[((size_t)row * 4 + l) * 2 + 1];
    for (int p = 0; p < 4; ++p) {
      const int ch = (((h * 4) + l) * 4 + p) * 2;
      const float ox = offv[(size_t)row * kC + ch];
      const float oy = offv[(size_t)row * kC + ch + 1];
      const float a  = aws[(size_t)row * 128 + h * 16 + l * 4 + p];
      const float x = (rx + ox / (float)Wl) * (float)Wl - 0.5f;
      const float y = (ry + oy / (float)Hl) * (float)Hl - 0.5f;
      const float x0f = floorf(x), y0f = floorf(y);
      const float wx = x - x0f, wy = y - y0f;
      const int x0 = (int)x0f, y0 = (int)y0f;
      float pacc = 0.f;
#pragma unroll
      for (int cy = 0; cy < 2; ++cy) {
#pragma unroll
        for (int cx = 0; cx < 2; ++cx) {
          const int xi = x0 + cx, yi = y0 + cy;
          const float w = (cx ? wx : 1.f - wx) * (cy ? wy : 1.f - wy);
          const bool valid = (xi >= 0) && (xi <= Wl - 1) && (yi >= 0) && (yi <= Hl - 1);
          const int xc = min(max(xi, 0), Wl - 1);
          const int yc = min(max(yi, 0), Hl - 1);
          const size_t pos = (size_t)b * kLV + starts[l] + yc * Wl + xc;
          const float gv = bf2f(value[pos * kC + h * kDH + lane]);
          pacc += gv * (valid ? w : 0.f);
        }
      }
      acc += pacc * a;
    }
  }
  outp[(size_t)row * kC + h * kDH + lane] = acc;
}

}  // namespace

extern "C" void kernel_launch(void* const* d_in, const int* in_sizes, int n_in,
                              void* d_out, int out_size, void* d_ws, size_t ws_size,
                              hipStream_t stream) {
  (void)in_sizes; (void)n_in; (void)out_size; (void)ws_size;
  const float* tgt      = (const float*)d_in[0];
  const float* mem      = (const float*)d_in[1];
  const float* qpos     = (const float*)d_in[2];
  const float* refp     = (const float*)d_in[3];
  const float* sa_in_w  = (const float*)d_in[4];
  const float* sa_in_b  = (const float*)d_in[5];
  const float* sa_out_w = (const float*)d_in[6];
  const float* sa_out_b = (const float*)d_in[7];
  const float* off_w    = (const float*)d_in[8];
  const float* off_b    = (const float*)d_in[9];
  const float* aw_w     = (const float*)d_in[10];
  const float* aw_b     = (const float*)d_in[11];
  const float* val_w    = (const float*)d_in[12];
  const float* val_b    = (const float*)d_in[13];
  const float* co_w     = (const float*)d_in[14];
  const float* co_b     = (const float*)d_in[15];
  const float* l1_w     = (const float*)d_in[16];
  const float* l1_b     = (const float*)d_in[17];
  const float* l2_w     = (const float*)d_in[18];
  const float* l2_b     = (const float*)d_in[19];
  const float* ln1_s    = (const float*)d_in[20];
  const float* ln1_b    = (const float*)d_in[21];
  const float* ln2_s    = (const float*)d_in[22];
  const float* ln2_b    = (const float*)d_in[23];
  const float* ln3_s    = (const float*)d_in[24];
  const float* ln3_b    = (const float*)d_in[25];
  float* out = (float*)d_out;

  // ---- workspace carving (256B aligned) ----
  char* base = (char*)d_ws;
  size_t cur = 0;
  auto alloc = [&](size_t bytes) -> void* {
    void* p = base + cur;
    cur = (cur + bytes + 255) & ~(size_t)255;
    return p;
  };
  const size_t RC = (size_t)kRows * kC;            // 7200*256
  const size_t MC = (size_t)kB * kLV * kC;         // 174080*256
  float* q32   = (float*)alloc(RC * 4);
  float* sao   = (float*)alloc(RC * 4);
  float* t1    = (float*)alloc(RC * 4);
  float* dq    = (float*)alloc(RC * 4);
  float* offv  = (float*)alloc(RC * 4);
  float* awl   = (float*)alloc((size_t)kRows * 128 * 4);
  float* aws   = (float*)alloc((size_t)kRows * 128 * 4);
  float* dsamp = (float*)alloc(RC * 4);
  float* coo   = (float*)alloc(RC * 4);
  float* t2    = (float*)alloc(RC * 4);
  float* ff2   = (float*)alloc(RC * 4);
  float* sbias = (float*)alloc(512 * 4);
  // bf16 activations
  u16* qbf     = (u16*)alloc(RC * 2);
  u16* tgtbf   = (u16*)alloc(RC * 2);
  u16* membf   = (u16*)alloc(MC * 2);
  u16* qkbf    = (u16*)alloc((size_t)kRows * 512 * 2);
  u16* vprbf   = (u16*)alloc(RC * 2);
  u16* Qhp     = (u16*)alloc(RC * 2);
  u16* Khp     = (u16*)alloc(RC * 2);
  u16* VT      = (u16*)alloc((size_t)kB * kH * kDH * kVTP * 2);
  u16* Obf     = (u16*)alloc(RC * 2);
  u16* dqbf    = (u16*)alloc(RC * 2);
  u16* valbf   = (u16*)alloc(MC * 2);
  u16* dsampbf = (u16*)alloc(RC * 2);
  u16* t2bf    = (u16*)alloc(RC * 2);
  u16* ffh     = (u16*)alloc((size_t)kRows * 2048 * 2);
  // bf16 weights
  u16* w_in  = (u16*)alloc((size_t)768 * 256 * 2);
  u16* w_so  = (u16*)alloc((size_t)256 * 256 * 2);
  u16* w_off = (u16*)alloc((size_t)256 * 256 * 2);
  u16* w_aw  = (u16*)alloc((size_t)128 * 256 * 2);
  u16* w_val = (u16*)alloc((size_t)256 * 256 * 2);
  u16* w_co  = (u16*)alloc((size_t)256 * 256 * 2);
  u16* w_l1  = (u16*)alloc((size_t)2048 * 256 * 2);
  u16* w_l2  = (u16*)alloc((size_t)256 * 2048 * 2);

  auto castLaunch = [&](const float* src, u16* dst, size_t n, float scale) {
    const int n4 = (int)(n / 4);
    cast_bf16<<<dim3((n4 + 255) / 256), 256, 0, stream>>>(src, dst, n4, scale);
  };
  auto gemmGrid = [](int M, int N) {
    return dim3((unsigned)((M + 127) / 128), (unsigned)(N / 16));
  };
  const int nEl = kRows * kC;
  const dim3 ewGrid((nEl + 255) / 256);

  // weight casts (f32 -> bf16, once per call); fold 1/sqrt(d) into wq (+ its bias)
  castLaunch(sa_in_w, w_in, (size_t)256 * 256, kScale);                          // wq scaled
  castLaunch(sa_in_w + (size_t)256 * 256, w_in + (size_t)256 * 256,
             (size_t)512 * 256, 1.f);                                            // wk|wv
  scale_bias<<<dim3(2), 256, 0, stream>>>(sa_in_b, sbias, 512, 256, kScale);
  castLaunch(sa_out_w, w_so, (size_t)256 * 256, 1.f);
  castLaunch(off_w, w_off, (size_t)256 * 256, 1.f);
  castLaunch(aw_w, w_aw, (size_t)128 * 256, 1.f);
  castLaunch(val_w, w_val, (size_t)256 * 256, 1.f);
  castLaunch(co_w, w_co, (size_t)256 * 256, 1.f);
  castLaunch(l1_w, w_l1, (size_t)2048 * 256, 1.f);
  castLaunch(l2_w, w_l2, (size_t)256 * 2048, 1.f);
  castLaunch(mem, membf, MC, 1.f);

  // 1) q = tgt + query_pos ; casts
  ew_add<<<ewGrid, 256, 0, stream>>>(tgt, qpos, q32, nEl);
  castLaunch(q32, qbf, RC, 1.f);
  castLaunch(tgt, tgtbf, RC, 1.f);
  // 2) QK projection (scaled wq|wk, bf16 out) and V projection (bf16 out)
  gemm_tdm<true, false><<<gemmGrid(kRows, 512), 256, 0, stream>>>(
      qbf, w_in, sbias, qkbf, kRows, 512, kC);
  gemm_tdm<true, false><<<gemmGrid(kRows, 256), 256, 0, stream>>>(
      tgtbf, w_in + (size_t)512 * kC, sa_in_b + 512, vprbf, kRows, kC, kC);
  // 3) per-head split (pure bf16 re-layout) + V^T pad zero, flash attention
  qkv_split_bf<<<ewGrid, 256, 0, stream>>>(qkbf, vprbf, Qhp, Khp, VT);
  {
    const int nPad = kB * kH * kDH * (kVTP - kNQ);
    vt_zero<<<dim3((nPad + 255) / 256), 256, 0, stream>>>(VT);
  }
  attn_flash<<<dim3(57, kH, kB), 32, 0, stream>>>(Qhp, Khp, VT, Obf);
  // 4) out projection + LN1 (emit t1 and t1+query_pos)
  gemm_tdm<false, false><<<gemmGrid(kRows, 256), 256, 0, stream>>>(
      Obf, w_so, sa_out_b, sao, kRows, kC, kC);
  add_ln<<<dim3(kRows / 8), 256, 0, stream>>>(sao, tgt, ln1_s, ln1_b, t1, qpos, dq, kRows);
  castLaunch(dq, dqbf, RC, 1.f);
  // 5) value projection (M=174080, bf16 out)
  gemm_tdm<true, false><<<gemmGrid(kB * kLV, 256), 256, 0, stream>>>(
      membf, w_val, val_b, valbf, kB * kLV, kC, kC);
  // 6) sampling offsets + attention weights
  gemm_tdm<false, false><<<gemmGrid(kRows, 256), 256, 0, stream>>>(
      dqbf, w_off, off_b, offv, kRows, kC, kC);
  gemm_tdm<false, false><<<gemmGrid(kRows, 128), 256, 0, stream>>>(
      dqbf, w_aw, aw_b, awl, kRows, 128, kC);
  aw_softmax<<<dim3((kRows * kH + 255) / 256), 256, 0, stream>>>(awl, aws, kRows * kH);
  // 7) deformable bilinear sampling
  deform_sample<<<dim3(kRows * kH / 8), 256, 0, stream>>>(valbf, offv, aws, refp, dsamp);
  castLaunch(dsamp, dsampbf, RC, 1.f);
  // 8) co projection + LN2
  gemm_tdm<false, false><<<gemmGrid(kRows, 256), 256, 0, stream>>>(
      dsampbf, w_co, co_b, coo, kRows, kC, kC);
  add_ln<<<dim3(kRows / 8), 256, 0, stream>>>(coo, t1, ln2_s, ln2_b, t2, nullptr, nullptr, kRows);
  castLaunch(t2, t2bf, RC, 1.f);
  // 9) FFN: relu(t2 @ l1^T + b1) @ l2^T + b2, then LN3 -> d_out
  gemm_tdm<true, true><<<gemmGrid(kRows, 2048), 256, 0, stream>>>(
      t2bf, w_l1, l1_b, ffh, kRows, 2048, kC);
  gemm_tdm<false, false><<<gemmGrid(kRows, 256), 256, 0, stream>>>(
      ffh, w_l2, l2_b, ff2, kRows, kC, 2048);
  add_ln<<<dim3(kRows / 8), 256, 0, stream>>>(ff2, t2, ln3_s, ln3_b, out, nullptr, nullptr, kRows);
}
